// MPNNWithHierarchicalNNConv_41291815584473
// MI455X (gfx1250) — compile-verified, matrix operand order checked by reading, not run on hardware
//
#include <hip/hip_runtime.h>
#include <hip/hip_fp16.h>
#include <cmath>

// ---------------------------------------------------------------------------
// Input index map (assumption: top-level inputs in setup_inputs() insertion
// order; the "params" pytree flattened JAX-style = dict keys sorted at every
// level, lists in order):
//  0 z[N] 1 bond_type[E] 2 x_clique[C] 3 edge_index[2,E] 4 node2clique[2,A]
//  5 clique_edge_index[2,CE] 6 batch[N] 7 clique_batch[C]
//  8 bond_emb[8,32] 9 clique_edge_emb[1,32] 10 clique_emb[1000,64]
// 11 clique_lin.b[32] 12 clique_lin.w[64,32]
// layers l in {0,1}, base = 13 + 16*l:
//   +0 c2n.b +1 c2n.w +2 cc.bias +3 cc.nn1.b +4 cc.nn1.w +5 cc.nn2.b
//   +6 cc.nn2.w +7 cc.root +8 n2c.b +9 n2c.w +10 nc.bias +11 nc.nn1.b
//   +12 nc.nn1.w +13 nc.nn2.b +14 nc.nn2.w +15 nc.root
// 45 node_emb[100,64] 46 node_lin.b 47 node_lin.w
// 48 red1.b[64] 49 red1.w[128,64] 50 red2.b[1] 51 red2.w[64,1]
// 52 s2s_clique.bh 53 .bi 54 .wh[32,128] 55 .wi[64,128]
// 56 s2s_node.bh 57 .bi 58 .wh 59 .wi
// ---------------------------------------------------------------------------

typedef __attribute__((ext_vector_type(16))) _Float16 v16h;
typedef __attribute__((ext_vector_type(8)))  _Float16 v8h;
typedef __attribute__((ext_vector_type(8)))  float    v8f;

#define WMMA_F16(A_, B_, C_) \
  __builtin_amdgcn_wmma_f32_16x16x32_f16(false, (A_), false, (B_), (short)0, (C_), false, false)

__device__ __forceinline__ void atomicMaxFloat(float* addr, float val) {
  unsigned int* ua = (unsigned int*)addr;
  unsigned int old = *ua;
  while (true) {
    float fo = __uint_as_float(old);
    if (fo >= val) break;
    unsigned int assumed = old;
    old = atomicCAS(ua, assumed, __float_as_uint(val));
    if (old == assumed) break;
  }
}

// A-fragment for 16-bit A(16x32): lane holds row, VGPR pair r -> K =
// {2r | 16+2(r-4)} + 8*half: two contiguous 8-element runs per lane,
// [8h, 8h+8) -> a[0..7] and [16+8h, 16+8h+8) -> a[8..15].

// f32 source (embedding tables): 4x b128 loads + converts (+ optional relu).
__device__ __forceinline__ v16h load_a_row_f32(const float* rowp, int half, int relu_in) {
  const float4* p0 = (const float4*)(rowp + 8 * half);
  const float4* p1 = (const float4*)(rowp + 16 + 8 * half);
  float4 f0 = p0[0], f1 = p0[1], f2 = p1[0], f3 = p1[1];
  float t[16] = {f0.x, f0.y, f0.z, f0.w, f1.x, f1.y, f1.z, f1.w,
                 f2.x, f2.y, f2.z, f2.w, f3.x, f3.y, f3.z, f3.w};
  v16h a;
#pragma unroll
  for (int i = 0; i < 16; ++i) {
    float v = t[i];
    if (relu_in) v = fmaxf(v, 0.f);
    a[i] = (_Float16)v;
  }
  return a;
}

// f16 mirror source (hot path): two b128 loads, zero converts.
__device__ __forceinline__ v16h load_a_row_h(const _Float16* rowp, int half) {
  v8h lo = *(const v8h*)(rowp + 8 * half);
  v8h hi = *(const v8h*)(rowp + 16 + 8 * half);
  v16h a;
#pragma unroll
  for (int i = 0; i < 8; ++i) { a[i] = lo[i]; a[8 + i] = hi[i]; }
  return a;
}

// ------------------------------- utility -----------------------------------
__global__ void mpnn_fill(float* p, float v, int n) {
  int i = blockIdx.x * 256 + threadIdx.x;
  if (i < n) p[i] = v;
}

// Precompute per-bond NNConv weight matrices: W_b = relu(ef_b@nn1+b1)@nn2+b2
__global__ void mpnn_prep_w(const float* __restrict__ emb, int relu_emb,
                            const float* __restrict__ nn1w, const float* __restrict__ nn1b,
                            const float* __restrict__ nn2w, const float* __restrict__ nn2b,
                            float* __restrict__ Wout) {
  int b = blockIdx.x;
  __shared__ float ef[32];
  __shared__ float h[32];
  int t = threadIdx.x;  // 256 threads
  if (t < 32) {
    float v = emb[b * 32 + t];
    if (relu_emb) v = fmaxf(v, 0.f);
    ef[t] = v;
  }
  __syncthreads();
  if (t < 32) {
    float s = nn1b[t];
    for (int k = 0; k < 32; ++k) s += ef[k] * nn1w[k * 32 + t];
    h[t] = fmaxf(s, 0.f);
  }
  __syncthreads();
  for (int o = t; o < 1024; o += 256) {
    float s = nn2b[o];
    for (int k = 0; k < 32; ++k) s += h[k] * nn2w[(size_t)k * 1024 + o];
    Wout[(size_t)b * 1024 + o] = s;
  }
}

// Re-pack W[nmat][Kdim][32] (f32, row-major) into WMMA B-fragment layout f16:
// out halves laid out [mat][kstep][tile][lane][16] so each lane's fragment is
// one contiguous 32-byte vector.
__global__ void mpnn_pack_w(const float* __restrict__ W, int Kdim, int nmat,
                            _Float16* __restrict__ out) {
  int t = blockIdx.x * 256 + threadIdx.x;
  int perMat = (Kdim >> 5) * 1024;
  int total = nmat * perMat;
  if (t >= total) return;
  int mat = t / perMat;
  int rem = t - mat * perMat;
  int ks = rem >> 10;
  int rem2 = rem & 1023;
  int tile = rem2 >> 9;
  int lane = (rem2 >> 4) & 31;
  int e = rem2 & 15;
  int r = e >> 1, odd = e & 1;
  int half = lane >> 4;
  int ncol = (lane & 15) + 16 * tile;
  int k0 = ((r < 4) ? 2 * r : 16 + 2 * (r - 4)) + 8 * half + ks * 32 + odd;
  out[t] = (_Float16)W[(size_t)mat * Kdim * 32 + (size_t)k0 * 32 + ncol];
}

// ------------------------- WMMA dense transforms ----------------------------
// Y[n,32] = act( gatherRows(src)[n,Kdim] @ W + bias ); Bp = packed fragments.
// A read from f16 mirror Xh when given, else from f32 X (with optional relu).
// Optional f16 mirror output Yh. 8 waves per block, 16 rows per wave.
__global__ void mpnn_gemm16(const float* __restrict__ X, const _Float16* __restrict__ Xh,
                            const int* __restrict__ gather, int Kdim,
                            const _Float16* __restrict__ Bp, const float* __restrict__ bias,
                            float* __restrict__ Y, _Float16* __restrict__ Yh,
                            int n, int relu_in, int relu_out) {
  int lane = threadIdx.x & 31;
  int wave = threadIdx.x >> 5;
  int half = lane >> 4, ncol = lane & 15;
  int row0 = (blockIdx.x * 8 + wave) * 16;
  if (row0 >= n) return;
  int m = min(row0 + ncol, n - 1);
  int src = gather ? gather[m] : m;
  const v16h* fr = (const v16h*)Bp;
  v8f c0 = {}; v8f c1 = {};
  for (int ks = 0; ks * 32 < Kdim; ++ks) {
    v16h a = Xh ? load_a_row_h(Xh + (size_t)src * Kdim + ks * 32, half)
                : load_a_row_f32(X + (size_t)src * Kdim + ks * 32, half, relu_in);
    v16h b0 = fr[(ks * 2 + 0) * 32 + lane];
    v16h b1 = fr[(ks * 2 + 1) * 32 + lane];
    c0 = WMMA_F16(a, b0, c0);
    c1 = WMMA_F16(a, b1, c1);
  }
#pragma unroll
  for (int r = 0; r < 8; ++r) {
    int mrow = row0 + r + 8 * half;   // C/D: VGPR r holds row r+8*half
    if (mrow < n) {
      float v0 = c0[r] + bias[ncol];
      float v1 = c1[r] + bias[ncol + 16];
      if (relu_out) { v0 = fmaxf(v0, 0.f); v1 = fmaxf(v1, 0.f); }
      Y[(size_t)mrow * 32 + ncol] = v0;
      Y[(size_t)mrow * 32 + ncol + 16] = v1;
      if (Yh) {
        Yh[(size_t)mrow * 32 + ncol] = (_Float16)v0;
        Yh[(size_t)mrow * 32 + ncol + 16] = (_Float16)v1;
      }
    }
  }
}

// Per-edge NNConv messages: 16 edges per wave, A = gathered f16 rows Xh[src].
// Packed per-bond W table staged in LDS once per 8-wave block; all bond
// variants computed with WMMA, lane-selected by each edge row's bond type.
__global__ void mpnn_edge_msg(const _Float16* __restrict__ Xh, const int* __restrict__ src,
                              const int* __restrict__ dst, const int* __restrict__ bond,
                              const _Float16* __restrict__ Wp, int nbonds,
                              float* __restrict__ acc, float* __restrict__ cnt, int nE) {
  __shared__ alignas(32) _Float16 sW[8 * 1024];   // [bond][tile][lane][16]
  int tid = threadIdx.x;                          // 256 threads
  int lane = tid & 31, wave = tid >> 5;
  int half = lane >> 4, ncol = lane & 15;
  {
    const float4* g4 = (const float4*)Wp;
    float4* l4 = (float4*)sW;
    for (int i = tid; i < nbonds * 128; i += 256) l4[i] = g4[i];
  }
  __syncthreads();
  const v16h* fr = (const v16h*)sW;
  int ngroups = (nE + 15) / 16;
  int gstride = gridDim.x * 8;
  for (int g = blockIdx.x * 8 + wave; g < ngroups; g += gstride) {
    int e0 = g * 16;
    int ea = min(e0 + ncol, nE - 1);
    v16h a = load_a_row_h(Xh + (size_t)src[ea] * 32, half);
    int rbond[8];
#pragma unroll
    for (int r = 0; r < 8; ++r) {
      int eid = min(e0 + r + 8 * half, nE - 1);  // uniform per half-wave
      rbond[r] = bond ? bond[eid] : 0;
    }
    float res0[8], res1[8];
#pragma unroll
    for (int r = 0; r < 8; ++r) { res0[r] = 0.f; res1[r] = 0.f; }
    for (int b = 0; b < nbonds; ++b) {
      v16h b0 = fr[(b * 2 + 0) * 32 + lane];
      v16h b1 = fr[(b * 2 + 1) * 32 + lane];
      v8f c0 = {}, c1 = {};
      c0 = WMMA_F16(a, b0, c0);
      c1 = WMMA_F16(a, b1, c1);
#pragma unroll
      for (int r = 0; r < 8; ++r)
        if (rbond[r] == b) { res0[r] = c0[r]; res1[r] = c1[r]; }
    }
#pragma unroll
    for (int r = 0; r < 8; ++r) {
      int eid = e0 + r + 8 * half;
      if (eid < nE) {
        int d = dst[eid];
        atomicAdd(&acc[(size_t)d * 32 + ncol], res0[r]);
        atomicAdd(&acc[(size_t)d * 32 + ncol + 16], res1[r]);
        if (ncol == 0) atomicAdd(&cnt[d], 1.0f);
      }
    }
  }
}

// out = relu(acc/max(cnt,1) + X@root + bias); X@root via WMMA (A from f16
// mirror, packed root).  Writes both f32 state and its f16 mirror in place.
__global__ void mpnn_conv_finalize(const _Float16* __restrict__ Xh,
                                   const _Float16* __restrict__ rootP,
                                   const float* __restrict__ bias, const float* __restrict__ acc,
                                   const float* __restrict__ cnt, float* __restrict__ Xout,
                                   _Float16* __restrict__ XhOut, int n) {
  int lane = threadIdx.x & 31;
  int wave = threadIdx.x >> 5;
  int half = lane >> 4, ncol = lane & 15;
  int row0 = (blockIdx.x * 8 + wave) * 16;
  if (row0 >= n) return;
  int m = min(row0 + ncol, n - 1);
  const v16h* fr = (const v16h*)rootP;
  v16h a = load_a_row_h(Xh + (size_t)m * 32, half);
  v16h b0 = fr[lane];
  v16h b1 = fr[32 + lane];
  v8f c0 = {}, c1 = {};
  c0 = WMMA_F16(a, b0, c0);
  c1 = WMMA_F16(a, b1, c1);
#pragma unroll
  for (int r = 0; r < 8; ++r) {
    int mrow = row0 + r + 8 * half;
    if (mrow < n) {
      float ct = fmaxf(cnt[mrow], 1.f);
      float v0 = acc[(size_t)mrow * 32 + ncol] / ct + c0[r] + bias[ncol];
      float v1 = acc[(size_t)mrow * 32 + ncol + 16] / ct + c1[r] + bias[ncol + 16];
      v0 = fmaxf(v0, 0.f); v1 = fmaxf(v1, 0.f);
      Xout[(size_t)mrow * 32 + ncol] = v0;
      Xout[(size_t)mrow * 32 + ncol + 16] = v1;
      XhOut[(size_t)mrow * 32 + ncol] = (_Float16)v0;
      XhOut[(size_t)mrow * 32 + ncol + 16] = (_Float16)v1;
    }
  }
}

// ---------------------- hierarchical scatter-mean ---------------------------
__global__ void mpnn_scatter_rows(const float* __restrict__ T, const int* __restrict__ sidx,
                                  const int* __restrict__ didx, float* __restrict__ acc,
                                  float* __restrict__ cnt, int nA) {
  int a = blockIdx.x;
  if (a >= nA) return;
  int lane = threadIdx.x;  // 32
  int s = sidx[a], d = didx[a];
  atomicAdd(&acc[(size_t)d * 32 + lane], T[(size_t)s * 32 + lane]);
  if (lane == 0) atomicAdd(&cnt[d], 1.f);
}

// X += relu(acc/max(cnt,1)); refresh f16 mirror.
__global__ void mpnn_add_mean(float* __restrict__ X, _Float16* __restrict__ Xh,
                              const float* __restrict__ acc,
                              const float* __restrict__ cnt, int n) {
  int i = blockIdx.x * 256 + threadIdx.x;
  if (i >= n * 32) return;
  int row = i >> 5;
  float v = acc[i] / fmaxf(cnt[row], 1.f);
  float nv = X[i] + fmaxf(v, 0.f);
  X[i] = nv;
  Xh[i] = (_Float16)nv;
}

// ------------------------------- Set2Set ------------------------------------
__global__ void mpnn_s2s_lstm(const float* __restrict__ q, const float* __restrict__ h_in,
                              const float* __restrict__ c_in, const float* __restrict__ wi,
                              const float* __restrict__ bi, const float* __restrict__ wh,
                              const float* __restrict__ bh, float* __restrict__ h_out,
                              float* __restrict__ c_out, int nB) {
  int t = blockIdx.x * 128 + threadIdx.x;
  if (t >= nB * 32) return;
  int g = t >> 5, u = t & 31;
  float gv[4];
  for (int j = 0; j < 4; ++j) {
    int col = j * 32 + u;                       // torch gate order i,f,g,o
    float s = bi[col] + bh[col];
    for (int k = 0; k < 64; ++k) s += q[g * 64 + k] * wi[k * 128 + col];
    for (int k = 0; k < 32; ++k) s += h_in[g * 32 + k] * wh[k * 128 + col];
    gv[j] = s;
  }
  float ig = 1.f / (1.f + expf(-gv[0]));
  float fg = 1.f / (1.f + expf(-gv[1]));
  float gg = tanhf(gv[2]);
  float og = 1.f / (1.f + expf(-gv[3]));
  float cn = fg * c_in[t] + ig * gg;
  c_out[t] = cn;
  h_out[t] = og * tanhf(cn);
}

__global__ void mpnn_s2s_energy(const float* __restrict__ X, const int* __restrict__ idx,
                                const float* __restrict__ h, float* __restrict__ e,
                                float* __restrict__ m, int n) {
  int v = blockIdx.x * 128 + threadIdx.x;
  if (v >= n) return;
  int g = idx[v];
  float s = 0.f;
  for (int k = 0; k < 32; ++k) s += X[(size_t)v * 32 + k] * h[g * 32 + k];
  e[v] = s;
  atomicMaxFloat(&m[g], s);
}

__global__ void mpnn_s2s_expsum(float* __restrict__ e, const int* __restrict__ idx,
                                const float* __restrict__ m, float* __restrict__ denom, int n) {
  int v = blockIdx.x * 128 + threadIdx.x;
  if (v >= n) return;
  int g = idx[v];
  float mg = m[g];
  if (!(mg >= -3e38f && mg <= 3e38f)) mg = 0.f;  // -inf (empty seg) / NaN -> 0
  float a = expf(e[v] - mg);
  e[v] = a;
  atomicAdd(&denom[g], a);
}

__global__ void mpnn_s2s_weighted(const float* __restrict__ X, const float* __restrict__ e,
                                  const int* __restrict__ idx, const float* __restrict__ denom,
                                  float* __restrict__ r, int n) {
  int v = blockIdx.x;
  if (v >= n) return;
  int lane = threadIdx.x;  // 32
  int g = idx[v];
  float a = e[v] / fmaxf(denom[g], 1e-16f);
  atomicAdd(&r[(size_t)g * 32 + lane], a * X[(size_t)v * 32 + lane]);
}

__global__ void mpnn_s2s_qstar(const float* __restrict__ h, const float* __restrict__ r,
                               float* __restrict__ q, int nB) {
  int t = blockIdx.x * 256 + threadIdx.x;
  if (t >= nB * 64) return;
  int g = t >> 6, u = t & 63;
  q[t] = (u < 32) ? h[g * 32 + u] : r[g * 32 + (u - 32)];
}

// ------------------------------- readout ------------------------------------
__global__ void mpnn_red1(const float* __restrict__ qn, const float* __restrict__ qc,
                          const float* __restrict__ w, const float* __restrict__ b,
                          float* __restrict__ hidden, int nB) {
  int t = blockIdx.x * 256 + threadIdx.x;
  if (t >= nB * 64) return;
  int g = t >> 6, u = t & 63;
  float s = b[u];
  for (int k = 0; k < 64; ++k) s += qn[g * 64 + k] * w[k * 64 + u];
  for (int k = 0; k < 64; ++k) s += qc[g * 64 + k] * w[(64 + k) * 64 + u];
  hidden[t] = fmaxf(s, 0.f);
}

__global__ void mpnn_red2(const float* __restrict__ hidden, const float* __restrict__ w,
                          const float* __restrict__ b, float* __restrict__ out, int nB) {
  int g = blockIdx.x * 128 + threadIdx.x;
  if (g >= nB) return;
  float s = b[0];
  for (int k = 0; k < 64; ++k) s += hidden[g * 64 + k] * w[k];
  out[g] = s;
}

// ------------------------------ host side -----------------------------------
static inline int hmin(int a, int b) { return a < b ? a : b; }

static void run_set2set(const float* X, const int* idx, int n,
                        const float* wi, const float* bi,
                        const float* wh, const float* bh,
                        float* q, float* hA, float* hB, float* cA, float* cB,
                        float* ebuf, float* m, float* denom, float* rbuf,
                        int nB, hipStream_t stream) {
  mpnn_fill<<<(nB * 64 + 255) / 256, 256, 0, stream>>>(q, 0.f, nB * 64);
  mpnn_fill<<<(nB * 32 + 255) / 256, 256, 0, stream>>>(hA, 0.f, nB * 32);
  mpnn_fill<<<(nB * 32 + 255) / 256, 256, 0, stream>>>(cA, 0.f, nB * 32);
  for (int s = 0; s < 3; ++s) {
    mpnn_s2s_lstm<<<(nB * 32 + 127) / 128, 128, 0, stream>>>(q, hA, cA, wi, bi, wh, bh, hB, cB, nB);
    mpnn_fill<<<(nB + 255) / 256, 256, 0, stream>>>(m, -INFINITY, nB);
    mpnn_fill<<<(nB + 255) / 256, 256, 0, stream>>>(denom, 0.f, nB);
    mpnn_fill<<<(nB * 32 + 255) / 256, 256, 0, stream>>>(rbuf, 0.f, nB * 32);
    mpnn_s2s_energy<<<(n + 127) / 128, 128, 0, stream>>>(X, idx, hB, ebuf, m, n);
    mpnn_s2s_expsum<<<(n + 127) / 128, 128, 0, stream>>>(ebuf, idx, m, denom, n);
    mpnn_s2s_weighted<<<n, 32, 0, stream>>>(X, ebuf, idx, denom, rbuf, n);
    mpnn_s2s_qstar<<<(nB * 64 + 255) / 256, 256, 0, stream>>>(hB, rbuf, q, nB);
    float* t = hA; hA = hB; hB = t;
    t = cA; cA = cB; cB = t;
  }
}

extern "C" void kernel_launch(void* const* d_in, const int* in_sizes, int n_in,
                              void* d_out, int out_size, void* d_ws, size_t ws_size,
                              hipStream_t stream) {
  const int* z      = (const int*)d_in[0];
  const int* bond   = (const int*)d_in[1];
  const int* xcl    = (const int*)d_in[2];
  const int* ei     = (const int*)d_in[3];
  const int* n2ci   = (const int*)d_in[4];
  const int* cei    = (const int*)d_in[5];
  const int* batch  = (const int*)d_in[6];
  const int* cbatch = (const int*)d_in[7];
  const int N = in_sizes[0], E = in_sizes[1], C = in_sizes[2];
  const int A = in_sizes[4] / 2, CE = in_sizes[5] / 2;
  const int nB = out_size;  // B graphs, output [B,1] f32
  const int* esrc = ei,   *edst = ei + E;
  const int* nidx = n2ci, *cidx = n2ci + A;
  const int* csrc = cei,  *cdst = cei + CE;

#define P(i) ((const float*)d_in[(i)])
  const float* bond_emb     = P(8);
  const float* cedge_emb    = P(9);
  const float* clique_emb   = P(10);
  const float* clique_lin_b = P(11);
  const float* clique_lin_w = P(12);
  const float* node_emb     = P(45);
  const float* node_lin_b   = P(46);
  const float* node_lin_w   = P(47);
  const float* red1_b = P(48), *red1_w = P(49);
  const float* red2_b = P(50), *red2_w = P(51);
  const float* s2sc_bh = P(52), *s2sc_bi = P(53), *s2sc_wh = P(54), *s2sc_wi = P(55);
  const float* s2sn_bh = P(56), *s2sn_bi = P(57), *s2sn_wh = P(58), *s2sn_wi = P(59);

  // -------- workspace carve --------
  char* wp = (char*)d_ws;
  auto carve = [&](size_t bytes) {
    char* p = wp;
    wp += (bytes + 255) & ~(size_t)255;
    return (void*)p;
  };
  float*    x   = (float*)carve((size_t)N * 32 * 4);
  float*    xc  = (float*)carve((size_t)C * 32 * 4);
  _Float16* xh  = (_Float16*)carve((size_t)N * 32 * 2);  // f16 mirror of x
  _Float16* xch = (_Float16*)carve((size_t)C * 32 * 2);  // f16 mirror of xc
  float* bufA = (float*)carve((size_t)N * 32 * 4);       // node-sized acc/tmp
  float* bufB = (float*)carve((size_t)C * 32 * 4);       // clique-sized acc/tmp
  float* cntN = (float*)carve((size_t)N * 4);
  float* cntC = (float*)carve((size_t)C * 4);
  float* Wtab = (float*)carve((size_t)2 * 9 * 1024 * 4);  // f32 bond tables
  // packed f16 fragment buffers
  _Float16* nodeLinP = (_Float16*)carve(2048 * 2);
  _Float16* cliqLinP = (_Float16*)carve(2048 * 2);
  _Float16* n2cP[2], *c2nP[2], *ncRootP[2], *ccRootP[2], *WtabNP[2], *WtabCP[2];
  for (int l = 0; l < 2; ++l) {
    n2cP[l]    = (_Float16*)carve(1024 * 2);
    c2nP[l]    = (_Float16*)carve(1024 * 2);
    ncRootP[l] = (_Float16*)carve(1024 * 2);
    ccRootP[l] = (_Float16*)carve(1024 * 2);
    WtabNP[l]  = (_Float16*)carve(8 * 1024 * 2);
    WtabCP[l]  = (_Float16*)carve(1024 * 2);
  }
  float* hA   = (float*)carve((size_t)nB * 32 * 4);
  float* hB   = (float*)carve((size_t)nB * 32 * 4);
  float* cA   = (float*)carve((size_t)nB * 32 * 4);
  float* cB   = (float*)carve((size_t)nB * 32 * 4);
  float* ebuf = (float*)carve((size_t)N * 4);
  float* mbuf = (float*)carve((size_t)nB * 4);
  float* dbuf = (float*)carve((size_t)nB * 4);
  float* rbuf = (float*)carve((size_t)nB * 32 * 4);
  float* qn   = (float*)carve((size_t)nB * 64 * 4);
  float* qc   = (float*)carve((size_t)nB * 64 * 4);
  float* hid  = (float*)carve((size_t)nB * 64 * 4);
  (void)ws_size; (void)n_in;

  // -------- precompute per-bond NNConv tables (the big algebraic win) ------
  for (int l = 0; l < 2; ++l) {
    int base = 13 + l * 16;
    mpnn_prep_w<<<8, 256, 0, stream>>>(bond_emb, 0, P(base + 12), P(base + 11),
                                       P(base + 14), P(base + 13),
                                       Wtab + (size_t)l * 9 * 1024);
    mpnn_prep_w<<<1, 256, 0, stream>>>(cedge_emb, 1, P(base + 4), P(base + 3),
                                       P(base + 6), P(base + 5),
                                       Wtab + (size_t)l * 9 * 1024 + 8 * 1024);
  }
  // -------- pack all GEMM weights into WMMA fragment layout ----------------
  mpnn_pack_w<<<8, 256, 0, stream>>>(node_lin_w, 64, 1, nodeLinP);
  mpnn_pack_w<<<8, 256, 0, stream>>>(clique_lin_w, 64, 1, cliqLinP);
  for (int l = 0; l < 2; ++l) {
    int base = 13 + l * 16;
    mpnn_pack_w<<<4, 256, 0, stream>>>(P(base + 9), 32, 1, n2cP[l]);
    mpnn_pack_w<<<4, 256, 0, stream>>>(P(base + 1), 32, 1, c2nP[l]);
    mpnn_pack_w<<<4, 256, 0, stream>>>(P(base + 15), 32, 1, ncRootP[l]);
    mpnn_pack_w<<<4, 256, 0, stream>>>(P(base + 7), 32, 1, ccRootP[l]);
    mpnn_pack_w<<<32, 256, 0, stream>>>(Wtab + (size_t)l * 9 * 1024, 32, 8, WtabNP[l]);
    mpnn_pack_w<<<4, 256, 0, stream>>>(Wtab + (size_t)l * 9 * 1024 + 8 * 1024, 32, 1, WtabCP[l]);
  }

  // -------- embeddings: x = relu(lin(relu(emb[idx]))) via WMMA -------------
  mpnn_gemm16<<<(N + 127) / 128, 256, 0, stream>>>(node_emb, nullptr, z, 64, nodeLinP,
                                                   node_lin_b, x, xh, N, 1, 1);
  mpnn_gemm16<<<(C + 127) / 128, 256, 0, stream>>>(clique_emb, nullptr, xcl, 64, cliqLinP,
                                                   clique_lin_b, xc, xch, C, 1, 1);

  // -------- message-passing layers -----------------------------------------
  for (int l = 0; l < 2; ++l) {
    int base = 13 + l * 16;
    const float* c2n_b   = P(base + 0);
    const float* cc_bias = P(base + 2);
    const float* n2c_b   = P(base + 8);
    const float* nc_bias = P(base + 10);

    // 1) node NNConv
    mpnn_fill<<<((N * 32) + 255) / 256, 256, 0, stream>>>(bufA, 0.f, N * 32);
    mpnn_fill<<<(N + 255) / 256, 256, 0, stream>>>(cntN, 0.f, N);
    mpnn_edge_msg<<<hmin(((E + 15) / 16 + 7) / 8, 512), 256, 0, stream>>>(
        xh, esrc, edst, bond, WtabNP[l], 8, bufA, cntN, E);
    mpnn_conv_finalize<<<(N + 127) / 128, 256, 0, stream>>>(xh, ncRootP[l], nc_bias,
                                                            bufA, cntN, x, xh, N);

    // 2) node -> clique: xc += relu(seg_mean(lin(x)[nidx], cidx))
    mpnn_gemm16<<<(N + 127) / 128, 256, 0, stream>>>(x, xh, nullptr, 32, n2cP[l], n2c_b,
                                                     bufA, (_Float16*)nullptr, N, 0, 0);
    mpnn_fill<<<((C * 32) + 255) / 256, 256, 0, stream>>>(bufB, 0.f, C * 32);
    mpnn_fill<<<(C + 255) / 256, 256, 0, stream>>>(cntC, 0.f, C);
    mpnn_scatter_rows<<<A, 32, 0, stream>>>(bufA, nidx, cidx, bufB, cntC, A);
    mpnn_add_mean<<<((C * 32) + 255) / 256, 256, 0, stream>>>(xc, xch, bufB, cntC, C);

    // 3) clique NNConv (single weight matrix)
    mpnn_fill<<<((C * 32) + 255) / 256, 256, 0, stream>>>(bufB, 0.f, C * 32);
    mpnn_fill<<<(C + 255) / 256, 256, 0, stream>>>(cntC, 0.f, C);
    mpnn_edge_msg<<<hmin(((CE + 15) / 16 + 7) / 8, 512), 256, 0, stream>>>(
        xch, csrc, cdst, nullptr, WtabCP[l], 1, bufB, cntC, CE);
    mpnn_conv_finalize<<<(C + 127) / 128, 256, 0, stream>>>(xch, ccRootP[l], cc_bias,
                                                            bufB, cntC, xc, xch, C);

    // 4) clique -> node: x += relu(seg_mean(lin(xc)[cidx], nidx))
    mpnn_gemm16<<<(C + 127) / 128, 256, 0, stream>>>(xc, xch, nullptr, 32, c2nP[l], c2n_b,
                                                     bufB, (_Float16*)nullptr, C, 0, 0);
    mpnn_fill<<<((N * 32) + 255) / 256, 256, 0, stream>>>(bufA, 0.f, N * 32);
    mpnn_fill<<<(N + 255) / 256, 256, 0, stream>>>(cntN, 0.f, N);
    mpnn_scatter_rows<<<A, 32, 0, stream>>>(bufB, cidx, nidx, bufA, cntN, A);
    mpnn_add_mean<<<((N * 32) + 255) / 256, 256, 0, stream>>>(x, xh, bufA, cntN, N);
  }

  // -------- Set2Set readouts (write q_star straight into qn / qc) ----------
  run_set2set(x, batch, N, s2sn_wi, s2sn_bi, s2sn_wh, s2sn_bh,
              qn, hA, hB, cA, cB, ebuf, mbuf, dbuf, rbuf, nB, stream);
  run_set2set(xc, cbatch, C, s2sc_wi, s2sc_bi, s2sc_wh, s2sc_bh,
              qc, hA, hB, cA, cB, ebuf, mbuf, dbuf, rbuf, nB, stream);

  // -------- final MLP -------------------------------------------------------
  mpnn_red1<<<(nB * 64 + 255) / 256, 256, 0, stream>>>(qn, qc, red1_w, red1_b, hid, nB);
  mpnn_red2<<<(nB + 127) / 128, 128, 0, stream>>>(hid, red2_w, red2_b, (float*)d_out, nB);
#undef P
}
